// CrossAttention_29678224015453
// MI455X (gfx1250) — compile-verified, matrix operand order checked by reading, not run on hardware
//
#include <hip/hip_runtime.h>
#include <hip/hip_bf16.h>

// ---------------------------------------------------------------------------
// Types / helpers
// ---------------------------------------------------------------------------
typedef _Float16 h16;
typedef __attribute__((ext_vector_type(16))) _Float16 v16h;
typedef __attribute__((ext_vector_type(8)))  float    v8f;

#define S_LEN 4096
#define CDIM  1280     // QUERY_DIM == INNER
#define DH    160
#define NHEAD 8
#define NB    2

union U4H8 { uint4 u4; unsigned int u[4]; h16 h[8]; };
union Frag { uint4 u4[2]; unsigned int u[8]; v16h v; };

static __device__ inline v8f vzero8() {
    v8f z = {0.f,0.f,0.f,0.f,0.f,0.f,0.f,0.f};
    return z;
}

static __device__ inline v8f wmma16(v16h a, v16h b, v8f c) {
    // D = A(16x32 f16) * B(32x16 f16) + C(16x16 f32)
    return __builtin_amdgcn_wmma_f32_16x16x32_f16(false, a, false, b, (short)0, c,
                                                  false, false);
}

// ---------------------------------------------------------------------------
// CDNA5 async / tensor data-mover plumbing (compile-guarded with fallbacks)
// ---------------------------------------------------------------------------
#if defined(__gfx1250__) && __has_builtin(__builtin_amdgcn_tensor_load_to_lds)
#define USE_TDM 1
typedef __attribute__((ext_vector_type(4))) unsigned int u32x4;
typedef __attribute__((ext_vector_type(8))) int          i32x8;
typedef __attribute__((ext_vector_type(4))) int          i32x4;

// 2D f16 tile load: global (row_stride elems between rows) -> LDS with row
// padding (pad_interval_code: 16 dwords->3, 64 dwords->5; pad_amount_code:
// N dwords -> N-1).  One TDM op; completion via TENSORcnt.
static __device__ inline void tdm_load_2d_f16(unsigned lds_addr, const void* gptr,
                                              unsigned tile_w, unsigned tile_h,
                                              unsigned long long row_stride,
                                              unsigned tensor_w, unsigned tensor_h,
                                              unsigned pad_interval_code,
                                              unsigned pad_amount_code) {
    const unsigned long long ga = (unsigned long long)(size_t)gptr;
    u32x4 g0;
    g0.x = 1u;                                              // count=1, user mode
    g0.y = lds_addr;                                        // LDS byte address
    g0.z = (unsigned)(ga & 0xffffffffull);                  // global addr lo
    g0.w = (unsigned)((ga >> 32) & 0x01ffffffull) | (2u << 30); // hi | type=2
    i32x8 g1;
    g1[0] = (int)((1u << 16)                                // data_size = 2B
                  | (1u << 20)                              // pad_enable
                  | (pad_interval_code << 22)
                  | (pad_amount_code << 25));
    g1[1] = (int)((tensor_w & 0xffffu) << 16);              // tensor_dim0 lo16
    g1[2] = (int)((tensor_w >> 16) | ((tensor_h & 0xffffu) << 16));
    g1[3] = (int)((tensor_h >> 16) | (tile_w << 16));       // tile_dim0
    g1[4] = (int)(tile_h & 0xffffu);                        // tile_dim1 (dim2=0)
    g1[5] = (int)(unsigned)(row_stride & 0xffffffffull);    // dim0_stride lo
    g1[6] = (int)(unsigned)((row_stride >> 32) & 0xffffull);
    g1[7] = 0;
    i32x4 z4 = {0, 0, 0, 0};
    i32x8 z8 = {0, 0, 0, 0, 0, 0, 0, 0};
    // clang-23 / therock arity: (g0, g1, g2, g3, g4, cpol)
    __builtin_amdgcn_tensor_load_to_lds(g0, g1, z4, z4, z8, 0);
}
#define TDM_WAIT() __builtin_amdgcn_s_wait_tensorcnt(0)
#else
#define USE_TDM 0
#endif

#if defined(__gfx1250__)
#define USE_ASYNC_LDS 1
// Per-lane async copy: one dword global -> LDS (ASYNCcnt tracked).
static __device__ inline void async_g2l_b32(unsigned lds_addr,
                                            const void* gptr) {
    asm volatile("global_load_async_to_lds_b32 %0, %1, off"
                 :: "v"(lds_addr), "v"((unsigned long long)(size_t)gptr)
                 : "memory");
}
#if __has_builtin(__builtin_amdgcn_s_wait_asynccnt)
#define ASYNC_WAIT() __builtin_amdgcn_s_wait_asynccnt(0)
#else
#define ASYNC_WAIT() asm volatile("s_wait_asynccnt 0x0" ::: "memory")
#endif
#else
#define USE_ASYNC_LDS 0
#endif

// A/B fragment from a row-major f16 LDS tile.  base_row = &tile[row][kbase].
// dword v holds K = (v>=4)*16 + (lane>=16)*8 + (v%4)*2  (and K+1).
static __device__ inline v16h frag_rowmajor(const h16* base_row, int lane) {
    const int hi8 = (lane & 16) ? 8 : 0;
    const uint4* p = (const uint4*)(base_row + hi8);
    Frag f;
    f.u4[0] = p[0];   // K = hi8 .. hi8+7
    f.u4[1] = p[2];   // K = hi8+16 .. hi8+23
    return f.v;
}

// A/B fragment from a K-pair-packed dword LDS tile: dword (kpair, col) holds
// elements (2*kpair, col) and (2*kpair+1, col).  basedw = &tile[kpair_base][0].
static __device__ inline v16h frag_packed(const unsigned int* basedw, int pitchdw,
                                          int col, int lane) {
    Frag f;
    const int kb = (lane & 16) ? 4 : 0;
#pragma unroll
    for (int v = 0; v < 8; ++v) {
        const int kp = kb + (v & 3) + ((v & 4) ? 8 : 0);
        f.u[v] = basedw[kp * pitchdw + col];
    }
    return f.v;
}

// Interleave two rows (k and k+1), 8 halves each, into 8 packed dwords in LDS.
static __device__ inline void pack_store(unsigned int* dst, uint4 q0, uint4 q1) {
    U4H8 a, b; a.u4 = q0; b.u4 = q1;
    unsigned int d[8];
#pragma unroll
    for (int j = 0; j < 8; ++j) {
        const unsigned int lo = (a.u[j >> 1] >> (16 * (j & 1))) & 0xffffu;
        const unsigned int hi = (b.u[j >> 1] >> (16 * (j & 1))) & 0xffffu;
        d[j] = lo | (hi << 16);
    }
    uint4* p = (uint4*)dst;
    p[0] = make_uint4(d[0], d[1], d[2], d[3]);
    p[1] = make_uint4(d[4], d[5], d[6], d[7]);
}

// ---------------------------------------------------------------------------
// fp32 -> f16 conversion
// ---------------------------------------------------------------------------
__global__ void cvt_f32_f16(const float* __restrict__ src, h16* __restrict__ dst,
                            size_t n) {
    size_t i = (size_t)blockIdx.x * blockDim.x + threadIdx.x;
    const size_t stride = (size_t)gridDim.x * blockDim.x;
    for (; i < n; i += stride) dst[i] = (h16)src[i];
}

// ---------------------------------------------------------------------------
// GEMM 1: Y[b][o][s] = sum_c W[o][c] * X[b][c][s]   (f16 in, f16 out)
// Block tile 128(M) x 128(N), K-step 32.  8 waves: wm in 0..3, wn in 0..1.
// A tile staged by TDM (wave 0); B tile pair-packed by all threads.
// ---------------------------------------------------------------------------
#define PA 40     // A LDS pitch (halves), mult of 8   (32 data + 4-dword pad)
#define PB 132    // packed-B LDS pitch (dwords), mult of 4

__global__ __launch_bounds__(256)
void gemm_xs(const h16* __restrict__ W, const h16* __restrict__ X,
             h16* __restrict__ Y) {
    __shared__ h16 Alds[128 * PA];
    __shared__ unsigned int Bp[16 * PB];

    const int b  = blockIdx.z;
    const int m0 = blockIdx.y * 128;
    const int n0 = blockIdx.x * 128;
    const h16* Xb = X + (size_t)b * CDIM * S_LEN;
    h16* Yb       = Y + (size_t)b * CDIM * S_LEN;

    const int tid  = threadIdx.x;
    const int lane = tid & 31;
    const int wv   = tid >> 5;
    const int wm   = wv >> 1;       // 0..3 -> 32 rows each
    const int wn   = wv & 1;        // 0..1 -> 64 cols each

    v8f acc[2][4];
#pragma unroll
    for (int i = 0; i < 2; ++i)
#pragma unroll
        for (int j = 0; j < 4; ++j) acc[i][j] = vzero8();

    for (int k0 = 0; k0 < CDIM; k0 += 32) {
        __syncthreads();
#if USE_TDM
        if (wv == 0) {
            // 128 rows x 32 halves, row stride CDIM, pitch 40 halves:
            // rows of 16 dwords (code 3) + 4-dword pad (code 3)
            tdm_load_2d_f16((unsigned)(size_t)(void*)Alds,
                            W + (size_t)m0 * CDIM + k0,
                            32, 128, (unsigned long long)CDIM,
                            CDIM, CDIM, 3, 3);
        }
#else
#pragma unroll
        for (int i = 0; i < 2; ++i) {
            const int id  = i * 256 + tid;
            const int row = id >> 2, seg = id & 3;
            *(uint4*)(Alds + row * PA + seg * 8) =
                *(const uint4*)(W + (size_t)(m0 + row) * CDIM + k0 + seg * 8);
        }
#endif
        // stage B packed: 16 kpairs x 128 cols (dwords)
        {
            const int kp = tid >> 4;   // 0..15
            const int sg = tid & 15;   // 0..15
            const uint4 r0 = *(const uint4*)(Xb + (size_t)(k0 + kp * 2)     * S_LEN + n0 + sg * 8);
            const uint4 r1 = *(const uint4*)(Xb + (size_t)(k0 + kp * 2 + 1) * S_LEN + n0 + sg * 8);
            pack_store(Bp + kp * PB + sg * 8, r0, r1);
        }
#if USE_TDM
        if (wv == 0) TDM_WAIT();
#endif
        __syncthreads();

        v16h bf[4];
#pragma unroll
        for (int ni = 0; ni < 4; ++ni)
            bf[ni] = frag_packed(Bp, PB, wn * 64 + ni * 16 + (lane & 15), lane);
#pragma unroll
        for (int mi = 0; mi < 2; ++mi) {
            const v16h a = frag_rowmajor(
                Alds + (wm * 32 + mi * 16 + (lane & 15)) * PA, lane);
#pragma unroll
            for (int ni = 0; ni < 4; ++ni)
                acc[mi][ni] = wmma16(a, bf[ni], acc[mi][ni]);
        }
    }

    // epilogue: f16 store, row o, col s
#pragma unroll
    for (int mi = 0; mi < 2; ++mi)
#pragma unroll
        for (int ni = 0; ni < 4; ++ni) {
            const int scol  = n0 + wn * 64 + ni * 16 + (lane & 15);
            const int rbase = m0 + wm * 32 + mi * 16 + ((lane & 16) ? 8 : 0);
#pragma unroll
            for (int r = 0; r < 8; ++r)
                Yb[(size_t)(rbase + r) * S_LEN + scol] = (h16)acc[mi][ni][r];
        }
}

// ---------------------------------------------------------------------------
// GEMM 2: Out[b][o][s] = sum_c W[o][c] * At[b][s][c] + bias[o]   (fp32 out)
// At is [S][1280] f16 => (c,c+1) dwords contiguous: per-lane async scatter
// copy global->LDS (ASYNCcnt); A tile via TDM.
// ---------------------------------------------------------------------------
__global__ __launch_bounds__(256)
void gemm_out(const h16* __restrict__ W, const h16* __restrict__ At,
              const float* __restrict__ bias, float* __restrict__ Out) {
    __shared__ h16 Alds[128 * PA];
    __shared__ unsigned int Bp[16 * PB];

    const int b  = blockIdx.z;
    const int m0 = blockIdx.y * 128;
    const int n0 = blockIdx.x * 128;
    const unsigned int* Atdw = (const unsigned int*)(At + (size_t)b * S_LEN * CDIM);
    float* Ob = Out + (size_t)b * CDIM * S_LEN;

    const int tid  = threadIdx.x;
    const int lane = tid & 31;
    const int wv   = tid >> 5;
    const int wm   = wv >> 1;
    const int wn   = wv & 1;

    v8f acc[2][4];
#pragma unroll
    for (int i = 0; i < 2; ++i)
#pragma unroll
        for (int j = 0; j < 4; ++j) acc[i][j] = vzero8();

    for (int k0 = 0; k0 < CDIM; k0 += 32) {
        __syncthreads();
#if USE_TDM
        if (wv == 0) {
            tdm_load_2d_f16((unsigned)(size_t)(void*)Alds,
                            W + (size_t)m0 * CDIM + k0,
                            32, 128, (unsigned long long)CDIM,
                            CDIM, CDIM, 3, 3);
        }
#else
#pragma unroll
        for (int i = 0; i < 2; ++i) {
            const int id  = i * 256 + tid;
            const int row = id >> 2, seg = id & 3;
            *(uint4*)(Alds + row * PA + seg * 8) =
                *(const uint4*)(W + (size_t)(m0 + row) * CDIM + k0 + seg * 8);
        }
#endif
        // stage B packed: dword scatter copy (16 kpairs x 128 cols)
        {
            const int sl = tid & 127;  // column s
            const int ch = tid >> 7;   // kpair half: 0..1
            const unsigned int* gsrc = Atdw + (size_t)(n0 + sl) * (CDIM / 2)
                                       + (k0 >> 1) + ch * 8;
#if USE_ASYNC_LDS
#pragma unroll
            for (int j = 0; j < 8; ++j)
                async_g2l_b32((unsigned)(size_t)(void*)(Bp + (ch * 8 + j) * PB + sl),
                              gsrc + j);
            ASYNC_WAIT();
#else
            U4H8 r0, r1;
            r0.u4 = *(const uint4*)(gsrc);
            r1.u4 = *(const uint4*)(gsrc + 4);
#pragma unroll
            for (int j = 0; j < 4; ++j) Bp[(ch * 8 + j) * PB + sl]     = r0.u[j];
#pragma unroll
            for (int j = 0; j < 4; ++j) Bp[(ch * 8 + 4 + j) * PB + sl] = r1.u[j];
#endif
        }
#if USE_TDM
        if (wv == 0) TDM_WAIT();
#endif
        __syncthreads();

        v16h bf[4];
#pragma unroll
        for (int ni = 0; ni < 4; ++ni)
            bf[ni] = frag_packed(Bp, PB, wn * 64 + ni * 16 + (lane & 15), lane);
#pragma unroll
        for (int mi = 0; mi < 2; ++mi) {
            const v16h a = frag_rowmajor(
                Alds + (wm * 32 + mi * 16 + (lane & 15)) * PA, lane);
#pragma unroll
            for (int ni = 0; ni < 4; ++ni)
                acc[mi][ni] = wmma16(a, bf[ni], acc[mi][ni]);
        }
    }

#pragma unroll
    for (int mi = 0; mi < 2; ++mi)
#pragma unroll
        for (int ni = 0; ni < 4; ++ni) {
            const int scol  = n0 + wn * 64 + ni * 16 + (lane & 15);
            const int rbase = m0 + wm * 32 + mi * 16 + ((lane & 16) ? 8 : 0);
#pragma unroll
            for (int r = 0; r < 8; ++r)
                Ob[(size_t)(rbase + r) * S_LEN + scol] =
                    acc[mi][ni][r] + bias[rbase + r];
        }
}

// ---------------------------------------------------------------------------
// Flash attention: one block per (q-tile=128, head, batch).
// scores = scale * Q^T K  (per head, d = 160),  online softmax,  O = P V^T.
// Q/K/V are f16 [b][1280][s].  Output At[b][s][1280] f16 (transposed).
// V tile staged by TDM; Q/K pair-packed by VALU.
// ---------------------------------------------------------------------------
#define PQK 132    // dword pitch of packed Qp/Kp/Pp tiles
#define PVH 136    // half pitch of V tile (128 data + 4-dword pad)

#define SM_QP   (80 * PQK)                       // dwords
#define SM_KP   (80 * PQK)
#define SM_VH   (160 * PVH)                      // halves
#define SM_PP   (64 * PQK)
#define ATTN_SMEM_BYTES ((SM_QP + SM_KP + SM_PP) * 4 + SM_VH * 2 + 256 * 4)

__global__ __launch_bounds__(256)
void attn_flash(const h16* __restrict__ Q, const h16* __restrict__ K,
                const h16* __restrict__ V, const float* __restrict__ mask,
                h16* __restrict__ At) {
    extern __shared__ char smem[];
    unsigned int* Qp = (unsigned int*)smem;           // [80][PQK] packed d-pairs x q
    unsigned int* Kp = Qp + SM_QP;                    // [80][PQK] packed d-pairs x k
    h16* Vl          = (h16*)(Kp + SM_KP);            // [160][PVH] row-major d x k
    unsigned int* Pp = (unsigned int*)(Vl + SM_VH);   // [64][PQK] packed k-pairs x q
    float* stats     = (float*)(Pp + SM_PP);          // [8][16] alpha per q
    float* lst       = stats + 128;                   // [8][16] l per q

    const int b  = blockIdx.z;
    const int hh = blockIdx.y;
    const int q0 = blockIdx.x * 128;
    const h16* Qb = Q + ((size_t)b * CDIM + hh * DH) * S_LEN;
    const h16* Kb = K + ((size_t)b * CDIM + hh * DH) * S_LEN;
    const h16* Vb = V + ((size_t)b * CDIM + hh * DH) * S_LEN;
    const float* mb = mask + (size_t)b * S_LEN;

    const int tid  = threadIdx.x;
    const int lane = tid & 31;
    const int wv   = tid >> 5;            // wave owns q rows wv*16 .. wv*16+15
    const float sc2 = 0.07905694150420948f * 1.4426950408889634f; // scale*log2(e)

    // stage Q tile once (packed d-pairs x 128 q columns)
    for (int t = tid; t < 80 * 16; t += 256) {
        const int dp = t >> 4, kg = t & 15;
        const uint4 r0 = *(const uint4*)(Qb + (size_t)(dp * 2)     * S_LEN + q0 + kg * 8);
        const uint4 r1 = *(const uint4*)(Qb + (size_t)(dp * 2 + 1) * S_LEN + q0 + kg * 8);
        pack_store(Qp + dp * PQK + kg * 8, r0, r1);
    }

    float m_r[8], l_r[8];
    v8f oacc[10];
#pragma unroll
    for (int r = 0; r < 8; ++r) { m_r[r] = -1e30f; l_r[r] = 0.f; }
#pragma unroll
    for (int t = 0; t < 10; ++t) oacc[t] = vzero8();

    for (int kt = 0; kt < S_LEN; kt += 128) {
        __syncthreads();
#if USE_TDM
        if (wv == 0) {
            // V tile: 160 rows x 128 halves, stride S_LEN, pitch 136 halves:
            // rows of 64 dwords (code 5) + 4-dword pad (code 3)
            tdm_load_2d_f16((unsigned)(size_t)(void*)Vl,
                            Vb + kt,
                            128, 160, (unsigned long long)S_LEN,
                            S_LEN, 160, 5, 3);
        }
#endif
        // stage K packed (+ V manually if no TDM)
        for (int t = tid; t < 80 * 16; t += 256) {
            const int dp = t >> 4, kg = t & 15;
            const uint4 r0 = *(const uint4*)(Kb + (size_t)(dp * 2)     * S_LEN + kt + kg * 8);
            const uint4 r1 = *(const uint4*)(Kb + (size_t)(dp * 2 + 1) * S_LEN + kt + kg * 8);
            pack_store(Kp + dp * PQK + kg * 8, r0, r1);
        }
#if !USE_TDM
        for (int t = tid; t < 160 * 16; t += 256) {
            const int d = t >> 4, kg = t & 15;
            *(uint4*)(Vl + d * PVH + kg * 8) =
                *(const uint4*)(Vb + (size_t)d * S_LEN + kt + kg * 8);
        }
#else
        if (wv == 0) TDM_WAIT();
#endif
        __syncthreads();

        // ---- scores: 16 q rows x 128 k cols per wave ----
        v8f sacc[8];
#pragma unroll
        for (int nb = 0; nb < 8; ++nb) sacc[nb] = vzero8();
#pragma unroll
        for (int dc = 0; dc < 5; ++dc) {
            const v16h a = frag_packed(Qp + dc * 16 * PQK, PQK,
                                       wv * 16 + (lane & 15), lane);
#pragma unroll
            for (int nb = 0; nb < 8; ++nb) {
                const v16h bf = frag_packed(Kp + dc * 16 * PQK, PQK,
                                            nb * 16 + (lane & 15), lane);
                sacc[nb] = wmma16(a, bf, sacc[nb]);
            }
        }
        // additive mask over k
#pragma unroll
        for (int nb = 0; nb < 8; ++nb) {
            const float mv = mb[kt + nb * 16 + (lane & 15)];
#pragma unroll
            for (int r = 0; r < 8; ++r) sacc[nb][r] += mv;
        }

        // ---- online softmax (rows 0..7 in lanes 0..15, rows 8..15 in 16..31) --
        float tm[8];
#pragma unroll
        for (int r = 0; r < 8; ++r) {
            tm[r] = sacc[0][r];
#pragma unroll
            for (int nb = 1; nb < 8; ++nb) tm[r] = fmaxf(tm[r], sacc[nb][r]);
        }
#pragma unroll
        for (int off = 1; off < 16; off <<= 1)
#pragma unroll
            for (int r = 0; r < 8; ++r)
                tm[r] = fmaxf(tm[r], __shfl_xor(tm[r], off, 32));

        float al[8], ts[8];
#pragma unroll
        for (int r = 0; r < 8; ++r) {
            const float mn = fmaxf(m_r[r], tm[r]);
            al[r] = exp2f((m_r[r] - mn) * sc2);
            m_r[r] = mn;
            ts[r] = 0.f;
        }
#pragma unroll
        for (int nb = 0; nb < 8; ++nb)
#pragma unroll
            for (int r = 0; r < 8; ++r) {
                const float p = exp2f((sacc[nb][r] - m_r[r]) * sc2);
                sacc[nb][r] = p;
                ts[r] += p;
            }
#pragma unroll
        for (int off = 1; off < 16; off <<= 1)
#pragma unroll
            for (int r = 0; r < 8; ++r) ts[r] += __shfl_xor(ts[r], off, 32);
#pragma unroll
        for (int r = 0; r < 8; ++r) l_r[r] = l_r[r] * al[r] + ts[r];

        // write P into packed [kpair][q] layout (own q columns only)
        {
            h16* Pph = (h16*)Pp;
            const int qrb = wv * 16 + ((lane & 16) ? 8 : 0);
#pragma unroll
            for (int nb = 0; nb < 8; ++nb)
#pragma unroll
                for (int r = 0; r < 8; ++r) {
                    const int kl = nb * 16 + (lane & 15);
                    Pph[((kl >> 1) * PQK + qrb + r) * 2 + (kl & 1)] =
                        (h16)sacc[nb][r];
                }
            // transpose alpha row-stats -> per-lane via LDS (wave-local, in-order)
            if ((lane & 15) == 0) {
                const int base = wv * 16 + ((lane & 16) ? 8 : 0);
#pragma unroll
                for (int r = 0; r < 8; ++r) stats[base + r] = al[r];
            }
        }
        const float aL = stats[wv * 16 + (lane & 15)];
#pragma unroll
        for (int t = 0; t < 10; ++t)
#pragma unroll
            for (int j = 0; j < 8; ++j) oacc[t][j] *= aL;

        // ---- O += V * P^T : M = d (10 blocks), N = q (own 16 cols) ----
#pragma unroll
        for (int dc = 0; dc < 4; ++dc) {
            const v16h bf = frag_packed(Pp + dc * 16 * PQK, PQK,
                                        wv * 16 + (lane & 15), lane);
#pragma unroll
            for (int mbi = 0; mbi < 10; ++mbi) {
                const v16h a = frag_rowmajor(
                    Vl + (mbi * 16 + (lane & 15)) * PVH + dc * 32, lane);
                oacc[mbi] = wmma16(a, bf, oacc[mbi]);
            }
        }
    }

    // ---- finalize: divide by l, store transposed At[b][q][h*160+d] ----
    if ((lane & 15) == 0) {
        const int base = wv * 16 + ((lane & 16) ? 8 : 0);
#pragma unroll
        for (int r = 0; r < 8; ++r) lst[base + r] = l_r[r];
    }
    const float lv = lst[wv * 16 + (lane & 15)];
    const float li = (lv > 0.f) ? (1.f / lv) : 0.f;
    const int qg = q0 + wv * 16 + (lane & 15);
    h16* Ab = At + (size_t)b * S_LEN * CDIM;
#pragma unroll
    for (int mbi = 0; mbi < 10; ++mbi) {
        U4H8 o;
#pragma unroll
        for (int j = 0; j < 8; ++j) o.h[j] = (h16)(oacc[mbi][j] * li);
        *(uint4*)(Ab + (size_t)qg * CDIM + hh * DH + mbi * 16 +
                  ((lane & 16) ? 8 : 0)) = o.u4;
    }
}

// ---------------------------------------------------------------------------
// Host launcher
// ---------------------------------------------------------------------------
extern "C" void kernel_launch(void* const* d_in, const int* in_sizes, int n_in,
                              void* d_out, int out_size, void* d_ws, size_t ws_size,
                              hipStream_t stream) {
    const float* hidden = (const float*)d_in[0];
    const float* ctx    = (const float*)d_in[1];
    const float* maskp  = (const float*)d_in[2];
    const float* Wq     = (const float*)d_in[3];
    const float* Wk     = (const float*)d_in[4];
    const float* Wv     = (const float*)d_in[5];
    const float* Wout   = (const float*)d_in[6];
    const float* bout   = (const float*)d_in[7];
    float* out = (float*)d_out;

    const size_t NW = (size_t)CDIM * CDIM;            // 1,638,400
    const size_t NX = (size_t)NB * CDIM * S_LEN;      // 10,485,760

    char* ws = (char*)d_ws;
    size_t off = 0;
    auto take = [&](size_t halves) { h16* p = (h16*)(ws + off); off += halves * sizeof(h16); return p; };
    h16* wq16 = take(NW);
    h16* wk16 = take(NW);
    h16* wv16 = take(NW);
    h16* wo16 = take(NW);
    h16* x16  = take(NX);
    h16* c16  = take(NX);
    h16* q16  = take(NX);
    h16* k16  = take(NX);
    h16* v16  = take(NX);
    h16* at16 = take(NX);   // transposed attention output [b][s][1280]
    (void)ws_size; (void)in_sizes; (void)n_in; (void)out_size;

    cvt_f32_f16<<<1024, 256, 0, stream>>>(Wq,   wq16, NW);
    cvt_f32_f16<<<1024, 256, 0, stream>>>(Wk,   wk16, NW);
    cvt_f32_f16<<<1024, 256, 0, stream>>>(Wv,   wv16, NW);
    cvt_f32_f16<<<1024, 256, 0, stream>>>(Wout, wo16, NW);
    cvt_f32_f16<<<4096, 256, 0, stream>>>(hidden, x16, NX);
    cvt_f32_f16<<<4096, 256, 0, stream>>>(ctx,    c16, NX);

    dim3 ggrid(S_LEN / 128, CDIM / 128, NB);   // (32, 10, 2)
    gemm_xs<<<ggrid, 256, 0, stream>>>(wq16, x16, q16);
    gemm_xs<<<ggrid, 256, 0, stream>>>(wk16, c16, k16);
    gemm_xs<<<ggrid, 256, 0, stream>>>(wv16, c16, v16);

    dim3 agrid(S_LEN / 128, NHEAD, NB);        // (32, 8, 2)
    attn_flash<<<agrid, 256, ATTN_SMEM_BYTES, stream>>>(q16, k16, v16, maskp, at16);

    gemm_out<<<ggrid, 256, 0, stream>>>(wo16, at16, bout, out);
}
